// BiLSTM_CRF_21053929685057
// MI455X (gfx1250) — matrix operand order repaired and checked by reading, not compiled, not run on hardware
//
#include <hip/hip_runtime.h>
#include <hip/hip_bf16.h>

// ---------------------------------------------------------------------------
// BiLSTM-CRF for MI455X (gfx1250, wave32, WMMA f32_16x16x32_f16)
//   V=30000 E=100 H=128 H2=256 K=9 B=128 T=512
// Pipeline:
//   k0 pack_weights : f32 -> f16 weight tiles (E padded to 128, K padded to 16)
//   k1 embed_gather : embedding[x] -> f16 activation matrix Ae[T*B][128]
//   k2 input_proj   : xp = Ae @ W_ih^T + b  (WMMA; stored in C-fragment layout)
//   k3 lstm_scan    : sequential scan, W_hh fragments pinned in VGPRs,
//                     h in LDS, c in VGPRs, WMMA per step; coalesced b128
//                     write-back of h_t via LDS staging (2 blocks: fwd/bwd)
//   k4 fc_emissions : emissions = [h_f|h_b] @ fc_W^T + fc_b   (WMMA, K=256)
//   k5 crf_loss     : CRF numerator + forward logsumexp, scalar output
// xp fragment layout: tile (s, mt, nt) -> xp[((s*8+mt)*32+nt)*256 + lane*8 + v]
// so producer stores and the scan's accumulator init are contiguous b128 ops.
// ---------------------------------------------------------------------------

typedef __attribute__((ext_vector_type(16))) _Float16 v16h;
typedef __attribute__((ext_vector_type(8)))  _Float16 v8h;
typedef __attribute__((ext_vector_type(8)))  float    v8f;
typedef __attribute__((ext_vector_type(4)))  float    v4f;

#define TT 512
#define BB 128
#define HH 128
#define G4 512   // 4*H
#define EP 128   // E padded

__device__ __forceinline__ v16h pack16(v8h lo, v8h hi) {
  v16h r;
#pragma unroll
  for (int i = 0; i < 8; ++i) { r[i] = lo[i]; r[8 + i] = hi[i]; }
  return r;
}

// A fragment (16x32 f16, M x K) from a row-major matrix.
// rowPtr = &M[row_of_this_lane][k0].  Contiguous k order (same as load_b_fragT):
//  lanes 0-15  hold M=lane,    elems 0-15: K=k0..k0+15
//  lanes 16-31 hold M=lane-16, elems 0-15: K=k0+16..k0+31
__device__ __forceinline__ v16h load_a_frag(const _Float16* rowPtr, int lane) {
  int kb = (lane >> 4) * 16;
  v8h lo = *(const v8h*)(rowPtr + kb);
  v8h hi = *(const v8h*)(rowPtr + kb + 8);
  return pack16(lo, hi);
}

// B fragment (32x16 f16, K x N) where B[k][n] = W[n][k], W row-major [N][ldw].
//  lane holds column n = n0 + lane%16; lanes 0-15: K=k0..k0+15, 16-31: +16
__device__ __forceinline__ v16h load_b_fragT(const _Float16* W, int ldw,
                                             int n0, int k0, int lane) {
  int n  = n0 + (lane & 15);
  int kb = (lane >> 4) * 16;
  const _Float16* p = W + (size_t)n * ldw + k0 + kb;
  v8h lo = *(const v8h*)(p);
  v8h hi = *(const v8h*)(p + 8);
  return pack16(lo, hi);
}

__device__ __forceinline__ v8f wmma32(v16h a, v16h b, v8f c) {
  return __builtin_amdgcn_wmma_f32_16x16x32_f16(false, a, false, b,
                                                (short)0, c, false, false);
}

__device__ __forceinline__ float sigmoidf_(float x) {
  return 1.f / (1.f + __expf(-x));
}

// ---------------------------------------------------------------- k0
__global__ __launch_bounds__(256) void pack_weights(
    const float* __restrict__ wihf, const float* __restrict__ whhf,
    const float* __restrict__ wihb, const float* __restrict__ whhb,
    const float* __restrict__ fcw,
    _Float16* __restrict__ wihf16, _Float16* __restrict__ whhf16,
    _Float16* __restrict__ wihb16, _Float16* __restrict__ whhb16,
    _Float16* __restrict__ fcw16) {
  int i = blockIdx.x * 256 + threadIdx.x;
  if (i < G4 * EP) {
    int r = i >> 7, c = i & 127;
    wihf16[i] = (_Float16)((c < 100) ? wihf[r * 100 + c] : 0.f);
    wihb16[i] = (_Float16)((c < 100) ? wihb[r * 100 + c] : 0.f);
    whhf16[i] = (_Float16)whhf[i];
    whhb16[i] = (_Float16)whhb[i];
  }
  if (i < 16 * 256) {
    int r = i >> 8, c = i & 255;
    fcw16[i] = (_Float16)((r < 9) ? fcw[r * 256 + c] : 0.f);
  }
}

// ---------------------------------------------------------------- k1
__global__ __launch_bounds__(128) void embed_gather(
    const int* __restrict__ x, const float* __restrict__ emb,
    _Float16* __restrict__ Ae) {
  int r = blockIdx.x;      // flattened (t,b): r = t*B + b
  int e = threadIdx.x;     // 0..127
  int t = r >> 7, b = r & 127;
  int id = x[b * TT + t];
  float v = (e < 100) ? emb[(size_t)id * 100 + e] : 0.f;
  Ae[(size_t)r * EP + e] = (_Float16)v;
}

// ---------------------------------------------------------------- k2
// xp = Ae @ W_ih^T + bias in C-fragment layout; dir b consumes reversed time.
__global__ __launch_bounds__(256) void input_proj(
    const _Float16* __restrict__ Ae,
    const _Float16* __restrict__ wihf, const _Float16* __restrict__ wihb,
    const float* __restrict__ bf, const float* __restrict__ bb,
    float* __restrict__ xpf, float* __restrict__ xpb) {
  const int lane = threadIdx.x & 31;
  const int wid  = threadIdx.x >> 5;
  const int mt   = blockIdx.x;               // 0..4095 row tile of T*B
  const int nt   = blockIdx.y * 8 + wid;     // 0..31 col tile of 4H
  const int dir  = blockIdx.z;
  const _Float16* W = dir ? wihb : wihf;
  const float* bias = dir ? bb : bf;
  float* xp = dir ? xpb : xpf;

  const int n0 = nt * 16;
  const int r0 = mt * 16;
  int r = r0 + (lane & 15);
  int src = r;
  if (dir) { int t = r >> 7; int b = r & 127; src = (TT - 1 - t) * BB + b; }
  const _Float16* arow = Ae + (size_t)src * EP;

  float bv = bias[n0 + (lane & 15)];
  v8f acc;
#pragma unroll
  for (int v = 0; v < 8; ++v) acc[v] = bv;

#pragma unroll
  for (int kk = 0; kk < 4; ++kk) {
    v16h a = load_a_frag(arow + kk * 32, lane);
    v16h b = load_b_fragT(W, EP, n0, kk * 32, lane);
    acc = wmma32(a, b, acc);
  }
  // fragment-major store: contiguous 32B per lane (2x global_store_b128)
  float* p = xp + (((size_t)mt * 32 + nt) * 256) + lane * 8;
  v4f lo, hi;
#pragma unroll
  for (int v = 0; v < 4; ++v) { lo[v] = acc[v]; hi[v] = acc[4 + v]; }
  *(v4f*)(p)     = lo;
  *(v4f*)(p + 4) = hi;
}

// ---------------------------------------------------------------- k3
// One block per direction.  512 threads = 16 waves.
// wave w: ht = w&7 (16 h-cols strip), m0 = w>>3; row tiles {m0, m0+2, m0+4, m0+6}.
// W_hh B-fragments (4 gates x 4 k-steps) pinned in VGPRs; h in LDS; c in VGPRs.
// h_t write-back: C-fragment -> LDS (transpose scatter), then coalesced b128
// copy LDS -> global (64B per thread per step).
__global__ __launch_bounds__(512) void lstm_scan(
    const _Float16* __restrict__ whhf16, const _Float16* __restrict__ whhb16,
    const float* __restrict__ xpf, const float* __restrict__ xpb,
    _Float16* __restrict__ hf, _Float16* __restrict__ hb) {
  const int dir = blockIdx.x;
  const _Float16* W = dir ? whhb16 : whhf16;
  const float* xp   = dir ? xpb : xpf;
  _Float16* hout    = dir ? hb : hf;

  __shared__ _Float16 s_h[BB * HH];   // 32 KB: h_{s-1}, f16 row-major [b][h]

  const int lane = threadIdx.x & 31;
  const int w    = threadIdx.x >> 5;   // 0..15
  const int ht   = w & 7;
  const int m0w  = w >> 3;             // 0 or 1

  for (int i = threadIdx.x; i < BB * HH; i += 512) s_h[i] = (_Float16)0;

  // Pin W_hh fragments: gate gi, k-step kk -> n0 = gi*128 + ht*16, k0 = kk*32
  v16h bfr[4][4];
#pragma unroll
  for (int gi = 0; gi < 4; ++gi)
#pragma unroll
    for (int kk = 0; kk < 4; ++kk)
      bfr[gi][kk] = load_b_fragT(W, HH, gi * HH + ht * 16, kk * 32, lane);

  v8f cst[4];                           // cell state, lives across all 512 steps
#pragma unroll
  for (int ti = 0; ti < 4; ++ti)
#pragma unroll
    for (int v = 0; v < 8; ++v) cst[ti][v] = 0.f;

  __syncthreads();

  const int n    = ht * 16 + (lane & 15);  // h column
  const int mrow = (lane >> 4) * 8;
  const int coff = threadIdx.x * 32;       // 64B chunk for coalesced copy

  for (int s = 0; s < TT; ++s) {
    const int tmap = dir ? (TT - 1 - s) : s;
    if (s + 1 < TT)  // stream this wave's next-step fragments (global_prefetch_b8)
      __builtin_prefetch(
          xp + (((size_t)(s + 1) * 8 + m0w) * 32 + ht) * 256 + lane * 8, 0, 1);

    v8f hres[4];
#pragma unroll
    for (int ti = 0; ti < 4; ++ti) {
      const int mt = m0w + 2 * ti;       // row tile within this step
      const int b0 = mt * 16;
      v8f g[4];
#pragma unroll
      for (int gi = 0; gi < 4; ++gi) {   // fragment-major: 2x b128 per gate
        const float* p =
            xp + (((size_t)s * 8 + mt) * 32 + gi * 8 + ht) * 256 + lane * 8;
        v4f lo = *(const v4f*)(p);
        v4f hi = *(const v4f*)(p + 4);
#pragma unroll
        for (int v = 0; v < 4; ++v) { g[gi][v] = lo[v]; g[gi][4 + v] = hi[v]; }
      }
#pragma unroll
      for (int kk = 0; kk < 4; ++kk) {
        v16h a = load_a_frag(&s_h[(b0 + (lane & 15)) * HH + kk * 32], lane);
#pragma unroll
        for (int gi = 0; gi < 4; ++gi) g[gi] = wmma32(a, bfr[gi][kk], g[gi]);
      }
#pragma unroll
      for (int v = 0; v < 8; ++v) {
        float iv = sigmoidf_(g[0][v]);
        float fv = sigmoidf_(g[1][v]);
        float gv = tanhf(g[2][v]);
        float ov = sigmoidf_(g[3][v]);
        float c  = fv * cst[ti][v] + iv * gv;
        cst[ti][v] = c;
        hres[ti][v] = ov * tanhf(c);
      }
    }

    __syncthreads();   // all waves finished reading h_{s-1}
#pragma unroll
    for (int ti = 0; ti < 4; ++ti) {   // fragment -> row-major transpose in LDS
      const int b0 = (m0w + 2 * ti) * 16;
#pragma unroll
      for (int v = 0; v < 8; ++v) {
        int row = b0 + mrow + v;
        s_h[row * HH + n] = (_Float16)hres[ti][v];
      }
    }
    __syncthreads();   // h_s complete in LDS

    // coalesced write-back: 64B contiguous per thread (b128 pairs)
    {
      _Float16* dst = hout + (size_t)tmap * (BB * HH) + coff;
      const _Float16* srcp = &s_h[coff];
#pragma unroll
      for (int q = 0; q < 4; ++q)
        *(v8h*)(dst + q * 8) = *(const v8h*)(srcp + q * 8);
    }
    // next step's compute only reads s_h; writes are gated by the next
    // post-compute barrier, and each wave's copy loads complete before it
    // signals that barrier (dscnt wait in __syncthreads).
  }
}

// ---------------------------------------------------------------- k4
__global__ __launch_bounds__(64) void fc_emissions(
    const _Float16* __restrict__ hf, const _Float16* __restrict__ hb,
    const _Float16* __restrict__ fcw16, const float* __restrict__ fcb,
    float* __restrict__ em) {
  const int lane = threadIdx.x & 31;
  const int wid  = threadIdx.x >> 5;
  const int mt   = blockIdx.x * 2 + wid;   // 0..4095
  const int r0   = mt * 16;
  const int n    = lane & 15;

  float bv = (n < 9) ? fcb[n] : 0.f;
  v8f acc;
#pragma unroll
  for (int v = 0; v < 8; ++v) acc[v] = bv;

  const _Float16* arf = hf + (size_t)(r0 + (lane & 15)) * HH;
  const _Float16* arb = hb + (size_t)(r0 + (lane & 15)) * HH;
#pragma unroll
  for (int kk = 0; kk < 4; ++kk) {   // first 128 of K: h_f
    v16h a = load_a_frag(arf + kk * 32, lane);
    v16h b = load_b_fragT(fcw16, 256, 0, kk * 32, lane);
    acc = wmma32(a, b, acc);
  }
#pragma unroll
  for (int kk = 0; kk < 4; ++kk) {   // second 128 of K: h_b
    v16h a = load_a_frag(arb + kk * 32, lane);
    v16h b = load_b_fragT(fcw16, 256, 0, 128 + kk * 32, lane);
    acc = wmma32(a, b, acc);
  }
  const int mrow = (lane >> 4) * 8;
#pragma unroll
  for (int v = 0; v < 8; ++v) {
    int r = r0 + mrow + v;
    int t = r >> 7, b = r & 127;
    em[((size_t)b * TT + t) * 16 + n] = acc[v];   // (B,T,Kpad16)
  }
}

// ---------------------------------------------------------------- k5
__global__ __launch_bounds__(128) void crf_loss(
    const float* __restrict__ em, const int* __restrict__ tags,
    const unsigned char* __restrict__ mask,
    const float* __restrict__ st, const float* __restrict__ en,
    const float* __restrict__ trans, float* __restrict__ out) {
  __shared__ float s_tr[81];
  __shared__ float s_red[128];
  const int tid = threadIdx.x;
  if (tid < 81) s_tr[tid] = trans[tid];
  __syncthreads();

  const int b = tid;
  const float* eb = em + (size_t)b * TT * 16;
  const int* tg = tags + (size_t)b * TT;
  const unsigned char* mk = mask + (size_t)b * TT;

  // numerator
  float num = st[tg[0]] + eb[tg[0]];
  int cnt = (mk[0] != 0);
  for (int t = 1; t < TT; ++t) {
    float mf = (mk[t] != 0) ? 1.f : 0.f;
    num += mf * (s_tr[tg[t - 1] * 9 + tg[t]] + eb[t * 16 + tg[t]]);
    cnt += (mk[t] != 0);
  }
  num += en[tg[cnt - 1]];

  // forward algorithm (denominator)
  float alpha[9];
#pragma unroll
  for (int k = 0; k < 9; ++k) alpha[k] = st[k] + eb[k];
  for (int t = 1; t < TT; ++t) {
    float nxt[9];
#pragma unroll
    for (int kp = 0; kp < 9; ++kp) {
      float mx = -1e30f;
#pragma unroll
      for (int k = 0; k < 9; ++k) mx = fmaxf(mx, alpha[k] + s_tr[k * 9 + kp]);
      float ss = 0.f;
#pragma unroll
      for (int k = 0; k < 9; ++k) ss += __expf(alpha[k] + s_tr[k * 9 + kp] - mx);
      nxt[kp] = mx + __logf(ss) + eb[t * 16 + kp];
    }
    if (mk[t] != 0) {
#pragma unroll
      for (int k = 0; k < 9; ++k) alpha[k] = nxt[k];
    }
  }
  float mx = -1e30f;
#pragma unroll
  for (int k = 0; k < 9; ++k) mx = fmaxf(mx, alpha[k] + en[k]);
  float ss = 0.f;
#pragma unroll
  for (int k = 0; k < 9; ++k) ss += __expf(alpha[k] + en[k] - mx);
  float denom = mx + __logf(ss);

  s_red[tid] = num - denom;
  __syncthreads();
  for (int off = 64; off > 0; off >>= 1) {
    if (tid < off) s_red[tid] += s_red[tid + off];
    __syncthreads();
  }
  if (tid == 0) out[0] = -s_red[0] / 128.f;
}

// ---------------------------------------------------------------- host
extern "C" void kernel_launch(void* const* d_in, const int* in_sizes, int n_in,
                              void* d_out, int out_size, void* d_ws, size_t ws_size,
                              hipStream_t stream) {
  const int*   x     = (const int*)  d_in[0];
  const int*   tags  = (const int*)  d_in[1];
  const unsigned char* mask = (const unsigned char*)d_in[2];
  const float* emb   = (const float*)d_in[3];
  const float* Wihf  = (const float*)d_in[4];
  const float* Whhf  = (const float*)d_in[5];
  const float* bf    = (const float*)d_in[6];
  const float* Wihb  = (const float*)d_in[7];
  const float* Whhb  = (const float*)d_in[8];
  const float* bb    = (const float*)d_in[9];
  const float* fcW   = (const float*)d_in[10];
  const float* fcb   = (const float*)d_in[11];
  const float* stt   = (const float*)d_in[12];
  const float* ent   = (const float*)d_in[13];
  const float* trans = (const float*)d_in[14];
  float* out = (float*)d_out;

  char* ws = (char*)d_ws;
  size_t o = 0;
  _Float16* Ae   = (_Float16*)(ws + o); o += (size_t)TT * BB * EP * 2;     // 16 MB
  float*    xpf  = (float*)   (ws + o); o += (size_t)TT * BB * G4 * 4;     // 128 MB
  float*    xpb  = (float*)   (ws + o); o += (size_t)TT * BB * G4 * 4;     // 128 MB
  _Float16* hf   = (_Float16*)(ws + o); o += (size_t)TT * BB * HH * 2;     // 16 MB
  _Float16* hb   = (_Float16*)(ws + o); o += (size_t)TT * BB * HH * 2;     // 16 MB
  float*    em   = (float*)   (ws + o); o += (size_t)TT * BB * 16 * 4;     // 4 MB
  _Float16* wihf16 = (_Float16*)(ws + o); o += (size_t)G4 * EP * 2;
  _Float16* wihb16 = (_Float16*)(ws + o); o += (size_t)G4 * EP * 2;
  _Float16* whhf16 = (_Float16*)(ws + o); o += (size_t)G4 * HH * 2;
  _Float16* whhb16 = (_Float16*)(ws + o); o += (size_t)G4 * HH * 2;
  _Float16* fcw16  = (_Float16*)(ws + o); o += (size_t)16 * 256 * 2;

  pack_weights<<<256, 256, 0, stream>>>(Wihf, Whhf, Wihb, Whhb, fcW,
                                        wihf16, whhf16, wihb16, whhb16, fcw16);
  embed_gather<<<TT * BB, 128, 0, stream>>>(x, emb, Ae);
  input_proj<<<dim3(4096, 4, 2), 256, 0, stream>>>(Ae, wihf16, wihb16, bf, bb,
                                                   xpf, xpb);
  lstm_scan<<<2, 512, 0, stream>>>(whhf16, whhb16, xpf, xpb, hf, hb);
  fc_emissions<<<2048, 64, 0, stream>>>(hf, hb, fcw16, fcb, em);
  crf_loss<<<1, 128, 0, stream>>>(em, tags, mask, stt, ent, trans, out);
}
